// MultiQueryAttention_11991548691116
// MI455X (gfx1250) — matrix operand order repaired and checked by reading, not compiled, not run on hardware
//
#include <hip/hip_runtime.h>

// ---------------- CDNA5 WMMA types ----------------
typedef __bf16 bh_t;
typedef __attribute__((ext_vector_type(16))) __bf16 v16bf;
typedef __attribute__((ext_vector_type(8)))  float  v8f;
typedef __attribute__((ext_vector_type(4)))  unsigned int v4u;

#define HIDDEN 1024
#define NHEADS 16
#define HDIM   64
#define BATCH  4
#define SEQ    1024
#define MTOT   (BATCH * SEQ)      // 4096
#define NBH    (BATCH * NHEADS)   // 64

// =====================================================================
// Tiled bf16-WMMA GEMM with bias:  C[M,N] = A[M,K] @ W[K,N] + bias[N]
// Block: 256 threads = 8 waves (4x2). Block tile 128x64, k-step 32.
// Each wave owns a 32x32 sub-tile -> 4 WMMAs per k-step from 2 A-frags
// and 2 B-frags (2x operand reuse, 4x WMMA density vs 16x16/wave).
// A is f32 or bf16 (template). Optional interleaved REPEAT-16 epilogue
// (MQA K/V broadcast) with packed 128-bit stores.
// =====================================================================
constexpr int BM = 128, BN = 64, BK = 32;
constexpr int AS_STR = BK + 2;   // LDS padding
constexpr int BS_STR = BK + 2;

template <bool A_BF16, bool OUT_BF16, int REPEAT>
__global__ __launch_bounds__(256) void gemm_bias(
    const void* __restrict__ Av, const float* __restrict__ W,
    const float* __restrict__ bias, void* __restrict__ Cv,
    int M, int N, int K) {
  __shared__ bh_t As[BM * AS_STR];           // 128x32 A tile, bf16 (~8.5KB)
  __shared__ bh_t Bs[BN * BS_STR];           // W tile transposed [n][k] (~4.3KB)

  const int tid  = threadIdx.x;
  const int lane = tid & 31;
  const int wid  = tid >> 5;
  const int wr   = wid >> 1;                 // 0..3 (M sub-tile, 32 rows)
  const int wc   = wid & 1;                  // 0..1 (N sub-tile, 32 cols)
  const int m0   = blockIdx.x * BM;
  const int n0   = blockIdx.y * BN;

  // staging assignments
  const int arow  = tid >> 1;                // 0..127
  const int akseg = (tid & 1) * 16;          // 0 or 16
  const int bkk   = tid >> 3;                // 0..31
  const int bnseg = (tid & 7) * 8;           // 0..56

  v8f acc[2][2] = {};

  for (int k0 = 0; k0 < K; k0 += BK) {
    // ---- stage A (convert f32->bf16 if needed), 16 elems/thread ----
    if (A_BF16) {
      const bh_t* A = (const bh_t*)Av;
      #pragma unroll
      for (int j = 0; j < 16; ++j)
        As[arow * AS_STR + akseg + j] = A[(size_t)(m0 + arow) * K + k0 + akseg + j];
    } else {
      const float* A = (const float*)Av;
      #pragma unroll
      for (int j = 0; j < 16; ++j)
        As[arow * AS_STR + akseg + j] = (bh_t)A[(size_t)(m0 + arow) * K + k0 + akseg + j];
    }
    // ---- stage W transposed: Bs[n][k], 8 elems/thread ----
    #pragma unroll
    for (int j = 0; j < 8; ++j)
      Bs[(bnseg + j) * BS_STR + bkk] = (bh_t)W[(size_t)(k0 + bkk) * N + n0 + bnseg + j];

    // prefetch next k-tile (global_prefetch_b8)
    if (k0 + BK < K) {
      if (A_BF16)
        __builtin_prefetch(&((const bh_t*)Av)[(size_t)(m0 + arow) * K + k0 + BK + akseg], 0, 1);
      else
        __builtin_prefetch(&((const float*)Av)[(size_t)(m0 + arow) * K + k0 + BK + akseg], 0, 1);
      __builtin_prefetch(&W[(size_t)(k0 + BK + bkk) * N + n0 + bnseg], 0, 1);
    }
    __syncthreads();

    const int row  = lane & 15;
    const int kb8  = (lane < 16) ? 0 : 8;
    const int kb16 = (lane < 16) ? 0 : 16;

    // ---- two A fragments (16x32, contiguous k = 16h + s) ----
    v16bf afrag[2];
    #pragma unroll
    for (int mi = 0; mi < 2; ++mi) {
      const bh_t* Asub = &As[(wr * 32 + mi * 16) * AS_STR];
      #pragma unroll
      for (int p = 0; p < 8; ++p) {
        const int kk = kb16 + 2 * p;
        afrag[mi][2 * p]     = Asub[row * AS_STR + kk];
        afrag[mi][2 * p + 1] = Asub[row * AS_STR + kk + 1];
      }
    }
    // ---- two B fragments (32x16, lane holds column n) ----
    v16bf bfrag[2];
    #pragma unroll
    for (int ni = 0; ni < 2; ++ni) {
      const bh_t* BsubT = &Bs[(wc * 32 + ni * 16) * BS_STR];
      #pragma unroll
      for (int p = 0; p < 8; ++p) {
        const int kk = kb16 + 2 * p;
        bfrag[ni][2 * p]     = BsubT[row * BS_STR + kk];
        bfrag[ni][2 * p + 1] = BsubT[row * BS_STR + kk + 1];
      }
    }
    // ---- 4 WMMAs per k-step ----
    #pragma unroll
    for (int mi = 0; mi < 2; ++mi)
      #pragma unroll
      for (int ni = 0; ni < 2; ++ni)
        acc[mi][ni] = __builtin_amdgcn_wmma_f32_16x16x32_bf16(
            false, afrag[mi], false, bfrag[ni], (short)0, acc[mi][ni], false, false);
    __syncthreads();
  }

  // ---- epilogue: bias + store ----
  const int n     = lane & 15;
  const int rbase = (lane < 16) ? 0 : 8;
  #pragma unroll
  for (int ni = 0; ni < 2; ++ni) {
    const int nglob = n0 + wc * 32 + ni * 16 + n;
    const float bvl = bias[nglob];
    #pragma unroll
    for (int mi = 0; mi < 2; ++mi) {
      #pragma unroll
      for (int v = 0; v < 8; ++v) {
        const int m = m0 + wr * 32 + mi * 16 + rbase + v;
        const float c = acc[mi][ni][v] + bvl;
        if (OUT_BF16) {
          bh_t* C = (bh_t*)Cv;
          if (REPEAT == 1) {
            C[(size_t)m * N + nglob] = (bh_t)c;
          } else {  // interleaved 16x repeat: two packed 128-bit stores
            const bh_t cb = (bh_t)c;
            unsigned short us;
            __builtin_memcpy(&us, &cb, 2);
            const unsigned int pk = (unsigned int)us | ((unsigned int)us << 16);
            const v4u q = {pk, pk, pk, pk};
            v4u* dst = (v4u*)&C[(size_t)m * N * REPEAT + nglob * REPEAT];
            dst[0] = q;
            dst[1] = q;
          }
        } else {
          ((float*)Cv)[(size_t)m * N + nglob] = c;
        }
      }
    }
  }
}

// =====================================================================
// Fused MQA attention per (bh, 16-row q-tile):
//   scores = Q(16x64) @ Krep^T / 32 * mask  -> LDS (f32, 16x1024)
//   full-row softmax (multiplicative-mask semantics, faithful)
//   attn written to d_out; ctx = attn @ Vrep -> bf16 workspace
// Block: 128 threads = 4 waves. Q fragments register-resident (reused
// across all 64 K-tiles).
// =====================================================================
__global__ __launch_bounds__(128) void mqa_attn(
    const bh_t* __restrict__ Q, const bh_t* __restrict__ Krep,
    const bh_t* __restrict__ Vrep, const float* __restrict__ mask,
    float* __restrict__ attn_out, bh_t* __restrict__ ctx) {
  __shared__ float sc[16 * SEQ];   // 64KB score tile; reused for ctx partials
  __shared__ float red[16][9];

  const int tid  = threadIdx.x;
  const int lane = tid & 31;
  const int w    = tid >> 5;                     // wave 0..3
  const int qt   = blockIdx.x;                   // 0..63
  const int bh   = blockIdx.y;                   // 0..63
  const size_t base = (size_t)bh * SEQ * HDIM;   // contiguous (bh,1024,64) slice
  const int q0 = qt * 16;

  const int row  = lane & 15;
  const int kb8  = (lane < 16) ? 0 : 8;
  const int kb16 = (lane < 16) ? 0 : 16;
  const int n    = lane & 15;

  // ---- Q A-fragments (16x64 = two 16x32), loaded once, reused 16x ----
  v16bf qa[2];
  #pragma unroll
  for (int h = 0; h < 2; ++h) {
    #pragma unroll
    for (int p = 0; p < 8; ++p) {
      const int kk = h * 32 + kb16 + 2 * p;
      qa[h][2 * p]     = Q[base + (size_t)(q0 + row) * HDIM + kk];
      qa[h][2 * p + 1] = Q[base + (size_t)(q0 + row) * HDIM + kk + 1];
    }
  }

  // ---- phase 1: scores = Q @ K^T * (1/sqrt(1024)) * mask ----
  const float scale = 0.03125f;  // 1/sqrt(HIDDEN)
  for (int kt = w * 16; kt < w * 16 + 16; ++kt) {
    v8f s = {};
    #pragma unroll
    for (int h = 0; h < 2; ++h) {
      v16bf kf;  // B = K^T (64x16): B[k][n] = Krep[kt*16+n][k], row-major reads
      #pragma unroll
      for (int p = 0; p < 8; ++p) {
        const int kk = h * 32 + kb16 + 2 * p;
        kf[2 * p]     = Krep[base + (size_t)(kt * 16 + n) * HDIM + kk];
        kf[2 * p + 1] = Krep[base + (size_t)(kt * 16 + n) * HDIM + kk + 1];
      }
      s = __builtin_amdgcn_wmma_f32_16x16x32_bf16(false, qa[h], false, kf,
                                                  (short)0, s, false, false);
    }
    const int rb = (lane < 16) ? 0 : 8;
    #pragma unroll
    for (int v = 0; v < 8; ++v) {
      const int qr   = rb + v;
      const int kcol = kt * 16 + n;
      const float mv = mask[(size_t)(q0 + qr) * SEQ + kcol];
      sc[qr * SEQ + kcol] = s[v] * scale * mv;
    }
  }
  __syncthreads();

  // ---- phase 2: softmax over full 1024-col row ----
  {
    const int r  = tid >> 3;          // 16 rows, 8 threads each
    const int c0 = (tid & 7) * 128;
    float mx = -3.4e38f;
    for (int c = 0; c < 128; ++c) mx = fmaxf(mx, sc[r * SEQ + c0 + c]);
    red[r][tid & 7] = mx;
    __syncthreads();
    float rmx = red[r][0];
    #pragma unroll
    for (int j = 1; j < 8; ++j) rmx = fmaxf(rmx, red[r][j]);
    // exp pass: cache exponentials in LDS (single transcendental per element)
    float lsum = 0.f;
    for (int c = 0; c < 128; ++c) {
      const float e = __expf(sc[r * SEQ + c0 + c] - rmx);
      sc[r * SEQ + c0 + c] = e;
      lsum += e;
    }
    __syncthreads();
    red[r][tid & 7] = lsum;
    __syncthreads();
    float rsum = 0.f;
    #pragma unroll
    for (int j = 0; j < 8; ++j) rsum += red[r][j];
    const float inv = 1.0f / rsum;
    float* arow = attn_out + ((size_t)bh * SEQ + (q0 + r)) * SEQ;
    for (int c = 0; c < 128; ++c) {
      const float p = sc[r * SEQ + c0 + c] * inv;
      sc[r * SEQ + c0 + c] = p;     // keep for attn @ V
      arow[c0 + c] = p;             // materialize attn output
    }
  }
  __syncthreads();

  // ---- phase 3: ctx(16x64) = attn(16x1024) @ V(1024x64) ----
  v8f cacc[4] = {};                  // four 16-col n-tiles
  for (int ks = w * 8; ks < w * 8 + 8; ++ks) {   // k-steps of 32
    v16bf af;                        // A = attn tile (f32 LDS -> bf16)
    #pragma unroll
    for (int p = 0; p < 8; ++p) {
      const int kk = ks * 32 + kb16 + 2 * p;
      af[2 * p]     = (bh_t)sc[row * SEQ + kk];
      af[2 * p + 1] = (bh_t)sc[row * SEQ + kk + 1];
    }
    #pragma unroll
    for (int nt = 0; nt < 4; ++nt) {
      v16bf vf;                      // B[k][n] = Vrep[ks*32+k][nt*16+n]
      #pragma unroll
      for (int p = 0; p < 8; ++p) {
        const int kk = ks * 32 + kb16 + 2 * p;
        vf[2 * p]     = Vrep[base + (size_t)kk * HDIM + nt * 16 + n];
        vf[2 * p + 1] = Vrep[base + (size_t)(kk + 1) * HDIM + nt * 16 + n];
      }
      cacc[nt] = __builtin_amdgcn_wmma_f32_16x16x32_bf16(false, af, false, vf,
                                                         (short)0, cacc[nt], false, false);
    }
  }
  __syncthreads();   // all waves done reading sc -> safe to reuse as partials
  {
    const int rb = (lane < 16) ? 0 : 8;
    #pragma unroll
    for (int nt = 0; nt < 4; ++nt)
      #pragma unroll
      for (int v = 0; v < 8; ++v)
        sc[w * 1024 + (rb + v) * HDIM + nt * 16 + n] = cacc[nt][v];
  }
  __syncthreads();
  // cross-wave reduction of 4 partials -> bf16 ctx
  for (int i = tid * 8; i < tid * 8 + 8; ++i) {
    const float s4 = sc[i] + sc[1024 + i] + sc[2048 + i] + sc[3072 + i];
    const int r = i >> 6, d = i & 63;
    ctx[base + (size_t)(q0 + r) * HDIM + d] = (bh_t)s4;
  }
}

// =====================================================================
extern "C" void kernel_launch(void* const* d_in, const int* in_sizes, int n_in,
                              void* d_out, int out_size, void* d_ws, size_t ws_size,
                              hipStream_t stream) {
  const float* inputs  = (const float*)d_in[0];
  const float* context = (const float*)d_in[1];
  const float* mask    = (const float*)d_in[2];
  const float* Wq = (const float*)d_in[3];
  const float* bq = (const float*)d_in[4];
  const float* Wk = (const float*)d_in[5];
  const float* bk = (const float*)d_in[6];
  const float* Wv = (const float*)d_in[7];
  const float* bv = (const float*)d_in[8];
  const float* Wo = (const float*)d_in[9];
  const float* bo = (const float*)d_in[10];

  float* out  = (float*)d_out;                        // (B,S,1024) f32
  float* attn = out + (size_t)MTOT * HIDDEN;          // (B,H,S,S)  f32

  // workspace layout (32 MB total): Q | Krep | Vrep | ctx  (bf16, 8 MB each)
  char* ws = (char*)d_ws;
  bh_t* Qb = (bh_t*)(ws);
  bh_t* Kb = (bh_t*)(ws + ((size_t)8 << 20));
  bh_t* Vb = (bh_t*)(ws + ((size_t)16 << 20));
  bh_t* Cx = (bh_t*)(ws + ((size_t)24 << 20));

  const dim3 blk(256);
  // Q projection: (4096x1024) = inputs @ Wq + bq  -> bf16
  gemm_bias<false, true, 1><<<dim3(MTOT / BM, HIDDEN / BN), blk, 0, stream>>>(
      inputs, Wq, bq, Qb, MTOT, HIDDEN, HIDDEN);
  // K/V projections: (4096x64) -> interleaved 16x repeat -> (4096x1024) bf16
  gemm_bias<false, true, 16><<<dim3(MTOT / BM, HDIM / BN), blk, 0, stream>>>(
      context, Wk, bk, Kb, MTOT, HDIM, HIDDEN);
  gemm_bias<false, true, 16><<<dim3(MTOT / BM, HDIM / BN), blk, 0, stream>>>(
      context, Wv, bv, Vb, MTOT, HDIM, HIDDEN);
  // fused attention (scores + softmax + attn@V), one block per (q-tile, bh)
  mqa_attn<<<dim3(SEQ / 16, NBH), dim3(128), 0, stream>>>(Qb, Kb, Vb, mask, attn, Cx);
  // output projection: ctx @ Wo + bo -> f32 outputs
  gemm_bias<true, false, 1><<<dim3(MTOT / BM, HIDDEN / BN), blk, 0, stream>>>(
      Cx, Wo, bo, out, MTOT, HIDDEN, HIDDEN);
}
